// ScoreNetwork_52948356825732
// MI455X (gfx1250) — matrix operand order repaired and checked by reading, not compiled, not run on hardware
//
#include <hip/hip_runtime.h>
#include <hip/hip_bf16.h>
#include <math.h>

// ---------------------------------------------------------------------------
// ScoreNetwork on MI455X (gfx1250, wave32).
// EdgeConv layers = batched GEMM [E,Kpad] x [Kpad,128] via v_wmma_f32_16x16x32_f16,
// one 16-edge x 128-col tile per wave, W kept as f16 in LDS (96KB of 320KB WGP LDS).
// All GEMM operands stream as f16 (HBM-bound: halves dominant traffic; values are
// identical to converting f32->f16 at the A-gather). A-fragments load as b128.
// Segment-max aggregation fused into the GEMM epilogue with u32-bit atomicMax
// (agg buffer == relu'd f32 node features of the next layer; f16 mirror for GEMM).
// SAGPool: GAT score (atomics) + 4-pass radix select top-k + masked edge remap.
// ---------------------------------------------------------------------------

typedef __attribute__((ext_vector_type(16))) _Float16 v16h;
typedef __attribute__((ext_vector_type(8)))  _Float16 v8h;
typedef __attribute__((ext_vector_type(8)))  float    v8f;

#define DEV __device__ __forceinline__

DEV unsigned okey(float f) {               // order-preserving float->u32 key
  unsigned u = __float_as_uint(f);
  return (u & 0x80000000u) ? ~u : (u | 0x80000000u);
}
DEV float odec(unsigned u) {
  return (u & 0x80000000u) ? __uint_as_float(u & 0x7fffffffu)
                           : __uint_as_float(~u);
}
DEV float lrelu(float x) { return x > 0.f ? x : 0.2f * x; }

// ---------------------------------------------------------------- fills
__global__ void fill_u32(unsigned* p, unsigned v, long long n) {
  long long i = (long long)blockIdx.x * blockDim.x + threadIdx.x;
  if (i < n) p[i] = v;
}
__global__ void fill_f32(float* p, float v, long long n) {
  long long i = (long long)blockIdx.x * blockDim.x + threadIdx.x;
  if (i < n) p[i] = v;
}
__global__ void cvt_f32h(const float* __restrict__ in, _Float16* __restrict__ out,
                         long long n) {
  long long i = (long long)blockIdx.x * blockDim.x + threadIdx.x;
  if (i < n) out[i] = (_Float16)in[i];
}

// ------------------------------------------------- input prep / W convert
__global__ void build_x0(const float* __restrict__ nf, const float* __restrict__ nl,
                         _Float16* __restrict__ x0, int N) {
  long long idx = (long long)blockIdx.x * blockDim.x + threadIdx.x;
  if (idx >= (long long)N * 20) return;
  int i = (int)(idx / 20), c = (int)(idx % 20);
  float v = (c < 16) ? nf[(long long)i * 16 + c] : nl[(long long)i * 4 + (c - 16)];
  x0[idx] = (_Float16)v;
}
// W [Kin,128] fp32 -> Wt [128,Kpad] f16 (transposed, zero-padded K)
__global__ void convert_wt(const float* __restrict__ W, _Float16* __restrict__ Wt,
                           int Kin, int Kpad) {
  int idx = blockIdx.x * blockDim.x + threadIdx.x;
  if (idx >= 128 * Kpad) return;
  int c = idx / Kpad, kk = idx % Kpad;
  float v = (kk < Kin) ? W[kk * 128 + c] : 0.f;
  Wt[idx] = (_Float16)v;
}

// -------------------------------------------------------- EdgeConv (WMMA)
// Per edge row: [xin[dst](xf), xin[src](xf), ein[e](ef), 0-pad] @ Wt^T + bias.
// e_out (f16) = emask ? relu(h) : 0 ; aggbits[dst] = atomic max of relu(h) bits.
// VEC=true requires xf==128 && ef==128 (8-half runs stay inside one segment and
// are 16B aligned -> single global_load_b128 per run).
template <bool VEC>
__global__ void __launch_bounds__(256) edgeconv_wmma(
    const _Float16* __restrict__ xin, int xf,
    const _Float16* __restrict__ ein, int ef,
    const int* __restrict__ src, const int* __restrict__ dst,
    const unsigned char* __restrict__ emask,
    const _Float16* __restrict__ Wt, int Kpad,
    const float* __restrict__ bias,
    _Float16* __restrict__ e_out,
    unsigned* __restrict__ aggbits,
    int E)
{
  extern __shared__ __align__(16) char dynsmem[];
  _Float16* sW = (_Float16*)dynsmem;
  const int tid = threadIdx.x;
  const int total = 128 * Kpad;
  for (int i = tid * 8; i < total; i += blockDim.x * 8)
    *(uint4*)(sW + i) = *(const uint4*)(Wt + i);
  __syncthreads();

  const int lane = tid & 31;
  const int wave = tid >> 5;
  const int wpb  = blockDim.x >> 5;
  const int gw   = blockIdx.x * wpb + wave;
  const int nw   = gridDim.x * wpb;
  const int ntiles = (E + 15) >> 4;

  const int r   = lane & 15;                 // A row / B,C column within tile
  const int g0  = (lane < 16) ? 0 : 8;       // A K-group base (ISA 16-bit A layout)
  const int kb  = (lane < 16) ? 0 : 8;       // B K-group base (matches A table order)
  const int rb  = (lane < 16) ? 0 : 8;       // C row base (ISA 32-bit C/D layout)
  const int KT  = Kpad >> 5;
  const int xf2 = xf * 2;

  for (int tile = gw; tile < ntiles; tile += nw) {
    const int e0 = tile << 4;
    // prefetch next tile's index cachelines (global_prefetch_b8, speculative)
    {
      int en = e0 + (nw << 4);
      if (en < E) {
        __builtin_prefetch(src + en, 0, 1);
        __builtin_prefetch(dst + en, 0, 1);
      }
    }
    const int ea = e0 + r;
    const int eac = (ea < E) ? ea : 0;
    const _Float16* pd = xin + (long long)dst[eac] * xf;
    const _Float16* ps = xin + (long long)src[eac] * xf;
    const _Float16* pe = ein + (long long)eac * ef;

    v8f acc[8];
    #pragma unroll
    for (int nt = 0; nt < 8; ++nt) {
      #pragma unroll
      for (int v = 0; v < 8; ++v) acc[nt][v] = 0.f;
    }

    for (int kc = 0; kc < KT; ++kc) {
      // ---- gather A fragment (16x32 f16): lane holds K=g0..g0+7, 16+g0..16+g0+7
      v16h a;
      if (VEC) {
        // xf == ef == 128: each 8-half run is one 16B load within one segment
        #pragma unroll
        for (int half = 0; half < 2; ++half) {
          const int kk = (kc << 5) + 16 * half + g0;
          const _Float16* q;
          if (kk < 128)      q = pd + kk;
          else if (kk < 256) q = ps + (kk - 128);
          else               q = pe + (kk - 256);
          const v8h run = *(const v8h*)(q);
          const int o = half * 8;
          #pragma unroll
          for (int t = 0; t < 8; ++t) a[o + t] = run[t];
        }
      } else {
        #pragma unroll
        for (int t = 0; t < 16; ++t) {
          const int kl = (t < 8) ? (g0 + t) : (16 + g0 + (t - 8));
          const int kk = (kc << 5) + kl;
          _Float16 v;
          if (kk < xf)            v = pd[kk];
          else if (kk < xf2)      v = ps[kk - xf];
          else if (kk < xf2 + ef) v = pe[kk - xf2];
          else                    v = (_Float16)0.f;
          a[t] = v;
        }
      }
      // ---- 8 column tiles: B fragment (32x16 f16) from LDS, contiguous per lane
      #pragma unroll
      for (int nt = 0; nt < 8; ++nt) {
        const _Float16* wb = sW + ((nt * 16 + r) * Kpad + (kc << 5) + kb);
        const v8h blo = *(const v8h*)(wb);
        const v8h bhi = *(const v8h*)(wb + 16);
        v16h b;
        #pragma unroll
        for (int t = 0; t < 8; ++t) { b[t] = blo[t]; b[8 + t] = bhi[t]; }
        acc[nt] = __builtin_amdgcn_wmma_f32_16x16x32_f16(
            false, a, false, b, (short)0, acc[nt], false, false);
      }
    }

    // ---- epilogue: bias + mask + relu, store f16 edge features, atomic-max agg
    int dr[8]; bool ok[8]; bool mv[8];
    #pragma unroll
    for (int v = 0; v < 8; ++v) {
      int e = e0 + rb + v;
      ok[v] = (e < E);
      int ec = ok[v] ? e : 0;
      dr[v] = dst[ec];
      mv[v] = ok[v] && (emask ? (emask[ec] != 0) : true);
    }
    #pragma unroll
    for (int nt = 0; nt < 8; ++nt) {
      const int col = nt * 16 + r;
      const float bc = bias[col];
      #pragma unroll
      for (int v = 0; v < 8; ++v) {
        float h  = acc[nt][v] + bc;
        float ho = (mv[v] && h > 0.f) ? h : 0.f;
        if (ok[v]) {
          e_out[(long long)(e0 + rb + v) * 128 + col] = (_Float16)ho;
          if (ho > 0.f)
            atomicMax(&aggbits[(long long)dr[v] * 128 + col], __float_as_uint(ho));
        }
      }
    }
  }
}

// ------------------------------------------------------------ GAT scoring
__global__ void gat_node1(const float* __restrict__ x, const float* __restrict__ gW,
                          const float* __restrict__ ga, float* __restrict__ h,
                          float* __restrict__ es, unsigned* __restrict__ mkey, int n) {
  int i = blockIdx.x * blockDim.x + threadIdx.x;
  if (i >= n) return;
  const float4* xr = (const float4*)(x + (long long)i * 128);
  const float4* gw = (const float4*)gW;
  float a = 0.f;
  #pragma unroll 8
  for (int c = 0; c < 32; ++c) {
    float4 xv = xr[c], wv = gw[c];
    a += xv.x * wv.x + xv.y * wv.y + xv.z * wv.z + xv.w * wv.w;
  }
  h[i] = a;
  float e = lrelu((ga[0] + ga[1]) * a);     // self-loop logit seeds segment max
  es[i] = e;
  mkey[i] = okey(e);
}
__global__ void gat_edge1(const int* __restrict__ s, const int* __restrict__ d,
                          const unsigned char* __restrict__ mi,
                          const float* __restrict__ h, const float* __restrict__ ga,
                          float* __restrict__ elog, unsigned* __restrict__ mkey, int E) {
  int e = blockIdx.x * blockDim.x + threadIdx.x;
  if (e >= E) return;
  bool m = mi ? (mi[e] != 0) : true;
  float v = 0.f;
  if (m) {
    v = lrelu(ga[0] * h[s[e]] + ga[1] * h[d[e]]);
    atomicMax(&mkey[d[e]], okey(v));
  }
  elog[e] = v;
}
__global__ void gat_node2(const float* __restrict__ h, const float* __restrict__ es,
                          const unsigned* __restrict__ mkey, float* __restrict__ m,
                          float* __restrict__ den, float* __restrict__ num, int n) {
  int i = blockIdx.x * blockDim.x + threadIdx.x;
  if (i >= n) return;
  float mm = odec(mkey[i]);
  float ws = expf(es[i] - mm);
  m[i] = mm; den[i] = ws; num[i] = ws * h[i];
}
__global__ void gat_edge2(const int* __restrict__ s, const int* __restrict__ d,
                          const unsigned char* __restrict__ mi,
                          const float* __restrict__ h, const float* __restrict__ elog,
                          const float* __restrict__ m, float* __restrict__ den,
                          float* __restrict__ num, int E) {
  int e = blockIdx.x * blockDim.x + threadIdx.x;
  if (e >= E) return;
  bool mk = mi ? (mi[e] != 0) : true;
  if (!mk) return;
  int di = d[e];
  float w = expf(elog[e] - m[di]);
  atomicAdd(&den[di], w);
  atomicAdd(&num[di], w * h[s[e]]);
}
__global__ void gat_node3(const float* __restrict__ num, const float* __restrict__ den,
                          const float* __restrict__ gb, float* __restrict__ score, int n) {
  int i = blockIdx.x * blockDim.x + threadIdx.x;
  if (i >= n) return;
  score[i] = num[i] / den[i] + gb[0];
}

// --------------------------------------------------- top-k (radix select)
// selst layout: [0]=prefix [1]=kneed [2]=cntG [3]=cntE [8..263]=hist
__global__ void sel_init(unsigned* st, unsigned k) {
  int t = threadIdx.x;
  if (t == 0) { st[0] = 0u; st[1] = k; st[2] = 0u; st[3] = 0u; }
  st[8 + t] = 0u;
}
__global__ void sel_hist(const float* __restrict__ sc, int n, unsigned* st, int b) {
  int i = blockIdx.x * blockDim.x + threadIdx.x;
  if (i >= n) return;
  unsigned key = okey(sc[i]);
  bool cand;
  if (b == 3) cand = true;
  else { int sh = (b + 1) * 8; cand = (key >> sh) == (st[0] >> sh); }
  if (cand) atomicAdd(&st[8 + ((key >> (b * 8)) & 255u)], 1u);
}
__global__ void sel_pick(unsigned* st, int b) {
  unsigned kneed = st[1], cum = 0; int chosen = 0;
  for (int bin = 255; bin >= 0; --bin) {
    unsigned c = st[8 + bin];
    if (cum + c >= kneed) { chosen = bin; break; }
    cum += c;
  }
  st[0] |= ((unsigned)chosen) << (b * 8);
  st[1] = kneed - cum;
  for (int i = 0; i < 256; ++i) st[8 + i] = 0u;
}
__global__ void sel_gt(const float* __restrict__ sc, int n, unsigned* st, int* perm) {
  int i = blockIdx.x * blockDim.x + threadIdx.x;
  if (i >= n) return;
  if (okey(sc[i]) > st[0]) { unsigned p = atomicAdd(&st[2], 1u); perm[p] = i; }
}
__global__ void sel_eq(const float* __restrict__ sc, int n, unsigned* st, int* perm, int k) {
  int i = blockIdx.x * blockDim.x + threadIdx.x;
  if (i >= n) return;
  if (okey(sc[i]) == st[0]) {
    unsigned q = atomicAdd(&st[3], 1u);
    unsigned p = st[2] + q;
    if (p < (unsigned)k) perm[p] = i;
  }
}

// ----------------------------------------------------- pooling / readout
// writes f32 (for GAT/readout) and f16 mirror (for the next conv's WMMA gather)
__global__ void pool_x(const float* __restrict__ x, const float* __restrict__ score,
                       const int* __restrict__ perm, float* __restrict__ out,
                       _Float16* __restrict__ outh, int k) {
  long long idx = (long long)blockIdx.x * blockDim.x + threadIdx.x;
  if (idx >= (long long)k * 128) return;
  int j = (int)(idx >> 7), c = (int)(idx & 127);
  int i = perm[j];
  float v = x[(long long)i * 128 + c] * tanhf(score[i]);
  out[idx] = v;
  outh[idx] = (_Float16)v;
}
__global__ void scatter_inv(const int* __restrict__ perm, int k, int* __restrict__ inv) {
  int j = blockIdx.x * blockDim.x + threadIdx.x;
  if (j < k) inv[perm[j]] = j;
}
__global__ void remap_edges(const int* __restrict__ s, const int* __restrict__ d,
                            const unsigned char* __restrict__ mi,
                            const int* __restrict__ inv, int* __restrict__ ns,
                            int* __restrict__ nd, unsigned char* __restrict__ mo, int E) {
  int e = blockIdx.x * blockDim.x + threadIdx.x;
  if (e >= E) return;
  bool base = mi ? (mi[e] != 0) : true;
  int is = inv[s[e]], id = inv[d[e]];
  bool m = base && (is >= 0) && (id >= 0);
  ns[e] = m ? is : 0;
  nd[e] = m ? id : 0;
  mo[e] = m ? 1 : 0;
}
__global__ void readout_acc(const float* __restrict__ xp, int k,
                            unsigned* __restrict__ rmax, float* __restrict__ rsum) {
  long long idx = (long long)blockIdx.x * blockDim.x + threadIdx.x;
  if (idx >= (long long)k * 128) return;
  int c = (int)(idx & 127);
  float v = xp[idx];
  atomicMax(&rmax[c], okey(v));
  atomicAdd(&rsum[c], v);
}
__global__ void readout_add(const unsigned* __restrict__ rmax,
                            const float* __restrict__ rsum, float k,
                            float* __restrict__ lbuf) {
  int c = threadIdx.x;
  lbuf[c]       += odec(rmax[c]);
  lbuf[c + 128] += rsum[c] / k;
}

// --------------------------------------------------------------- head MLP
__global__ void mlp_head(const float* __restrict__ l,
                         const float* __restrict__ lw1, const float* __restrict__ lb1,
                         const float* __restrict__ lw2, const float* __restrict__ lb2,
                         const float* __restrict__ lw3, const float* __restrict__ lb3,
                         float* __restrict__ out) {
  __shared__ float sl[256], h1[128], h2[64];
  int t = threadIdx.x;
  if (t < 256) sl[t] = l[t];
  __syncthreads();
  if (t < 128) {
    float a = lb1[t];
    for (int j = 0; j < 256; ++j) a += sl[j] * lw1[j * 128 + t];
    h1[t] = fmaxf(a, 0.f);
  }
  __syncthreads();
  if (t < 64) {
    float a = lb2[t];
    for (int j = 0; j < 128; ++j) a += h1[j] * lw2[j * 64 + t];
    h2[t] = fmaxf(a, 0.f);
  }
  __syncthreads();
  if (t < 4) {
    float a = lb3[t];
    for (int j = 0; j < 64; ++j) a += h2[j] * lw3[j * 4 + t];
    out[t] = a;
  }
}

// ---------------------------------------------------------------------------
static inline unsigned gdiv(long long n) { return (unsigned)((n + 255) / 256); }

extern "C" void kernel_launch(void* const* d_in, const int* in_sizes, int n_in,
                              void* d_out, int out_size, void* d_ws, size_t ws_size,
                              hipStream_t stream) {
  (void)n_in; (void)out_size; (void)ws_size;

  const float* node_feat  = (const float*)d_in[0];
  const float* node_level = (const float*)d_in[1];
  const int*   eidx       = (const int*)d_in[2];
  const float* edge_feat  = (const float*)d_in[3];
  const float* W1 = (const float*)d_in[4];  const float* b1 = (const float*)d_in[5];
  const float* W2 = (const float*)d_in[6];  const float* b2 = (const float*)d_in[7];
  const float* W3 = (const float*)d_in[8];  const float* b3 = (const float*)d_in[9];
  const float* W4 = (const float*)d_in[10]; const float* b4 = (const float*)d_in[11];
  const float* gW2 = (const float*)d_in[12]; const float* ga2 = (const float*)d_in[13]; const float* gb2 = (const float*)d_in[14];
  const float* gW3 = (const float*)d_in[15]; const float* ga3 = (const float*)d_in[16]; const float* gb3 = (const float*)d_in[17];
  const float* gW4 = (const float*)d_in[18]; const float* ga4 = (const float*)d_in[19]; const float* gb4 = (const float*)d_in[20];
  const float* lw1 = (const float*)d_in[21]; const float* lb1 = (const float*)d_in[22];
  const float* lw2 = (const float*)d_in[23]; const float* lb2 = (const float*)d_in[24];
  const float* lw3 = (const float*)d_in[25]; const float* lb3 = (const float*)d_in[26];

  const int N = in_sizes[0] / 16;       // node_feat [N,4,4]
  const int E = in_sizes[2] / 2;        // edge_index [2,E]
  const int* src = eidx;
  const int* dst = eidx + E;
  const int k1 = (N + 1) / 2, k2 = (k1 + 1) / 2, k3 = (k2 + 1) / 2;

  // ---- workspace carve-out (bump allocator, 256B aligned)
  char* wp = (char*)d_ws;
  auto alloc = [&](size_t bytes) -> void* {
    void* p = (void*)wp;
    wp += (bytes + 255) & ~(size_t)255;
    return p;
  };
  _Float16* x0h  = (_Float16*)alloc((size_t)N * 20 * 2);
  _Float16* efh  = (_Float16*)alloc((size_t)E * 4 * 2);
  _Float16* Wt1  = (_Float16*)alloc((size_t)128 * 64 * 2);
  _Float16* Wt2  = (_Float16*)alloc((size_t)128 * 384 * 2);
  _Float16* Wt3  = (_Float16*)alloc((size_t)128 * 384 * 2);
  _Float16* Wt4  = (_Float16*)alloc((size_t)128 * 384 * 2);
  _Float16* eA   = (_Float16*)alloc((size_t)E * 128 * 2);   // e1, later e3 (f16)
  _Float16* eB   = (_Float16*)alloc((size_t)E * 128 * 2);   // e2, later e4 (f16)
  unsigned* x1u  = (unsigned*)alloc((size_t)N * 128 * 4);
  unsigned* x2u  = (unsigned*)alloc((size_t)N * 128 * 4);
  unsigned* x3u  = (unsigned*)alloc((size_t)k1 * 128 * 4);
  unsigned* x4u  = (unsigned*)alloc((size_t)k2 * 128 * 4);
  _Float16* x1h  = (_Float16*)alloc((size_t)N * 128 * 2);   // f16 mirror of x1
  float*    xp1  = (float*)alloc((size_t)k1 * 128 * 4);
  float*    xp2  = (float*)alloc((size_t)k2 * 128 * 4);
  float*    xp3  = (float*)alloc((size_t)k3 * 128 * 4);
  _Float16* xp1h = (_Float16*)alloc((size_t)k1 * 128 * 2);
  _Float16* xp2h = (_Float16*)alloc((size_t)k2 * 128 * 2);
  _Float16* xp3h = (_Float16*)alloc((size_t)k3 * 128 * 2);
  float*    hA   = (float*)alloc((size_t)N * 4);
  float*    esA  = (float*)alloc((size_t)N * 4);
  float*    mmA  = (float*)alloc((size_t)N * 4);
  float*    denA = (float*)alloc((size_t)N * 4);
  float*    numA = (float*)alloc((size_t)N * 4);
  float*    scoreA = (float*)alloc((size_t)N * 4);
  unsigned* mkeyA  = (unsigned*)alloc((size_t)N * 4);
  float*    elogA  = (float*)alloc((size_t)E * 4);
  int*      perm   = (int*)alloc((size_t)k1 * 4);
  int*      invA   = (int*)alloc((size_t)N * 4);
  int*      s2 = (int*)alloc((size_t)E * 4);
  int*      d2 = (int*)alloc((size_t)E * 4);
  int*      s3 = (int*)alloc((size_t)E * 4);
  int*      d3 = (int*)alloc((size_t)E * 4);
  unsigned char* mask2 = (unsigned char*)alloc((size_t)E);
  unsigned char* mask3 = (unsigned char*)alloc((size_t)E);
  unsigned* selst = (unsigned*)alloc(264 * 4);
  unsigned* rmax  = (unsigned*)alloc(128 * 4);
  float*    rsum  = (float*)alloc(128 * 4);
  float*    lbuf  = (float*)alloc(256 * 4);

  // ---- prep
  build_x0<<<gdiv((long long)N * 20), 256, 0, stream>>>(node_feat, node_level, x0h, N);
  cvt_f32h<<<gdiv((long long)E * 4), 256, 0, stream>>>(edge_feat, efh, (long long)E * 4);
  convert_wt<<<gdiv(128 * 64),  256, 0, stream>>>(W1, Wt1, 44, 64);
  convert_wt<<<gdiv(128 * 384), 256, 0, stream>>>(W2, Wt2, 384, 384);
  convert_wt<<<gdiv(128 * 384), 256, 0, stream>>>(W3, Wt3, 384, 384);
  convert_wt<<<gdiv(128 * 384), 256, 0, stream>>>(W4, Wt4, 384, 384);
  fill_f32<<<1, 256, 0, stream>>>(lbuf, 0.f, 256);

  auto run_pool = [&](const float* x, int n, int k,
                      const int* ps, const int* pd, const unsigned char* pm,
                      const float* gW, const float* ga, const float* gb,
                      float* xp, _Float16* xph, int* ns, int* nd, unsigned char* mo) {
    gat_node1<<<gdiv(n), 256, 0, stream>>>(x, gW, ga, hA, esA, mkeyA, n);
    gat_edge1<<<gdiv(E), 256, 0, stream>>>(ps, pd, pm, hA, ga, elogA, mkeyA, E);
    gat_node2<<<gdiv(n), 256, 0, stream>>>(hA, esA, mkeyA, mmA, denA, numA, n);
    gat_edge2<<<gdiv(E), 256, 0, stream>>>(ps, pd, pm, hA, elogA, mmA, denA, numA, E);
    gat_node3<<<gdiv(n), 256, 0, stream>>>(numA, denA, gb, scoreA, n);
    sel_init<<<1, 256, 0, stream>>>(selst, (unsigned)k);
    for (int b = 3; b >= 0; --b) {
      sel_hist<<<gdiv(n), 256, 0, stream>>>(scoreA, n, selst, b);
      sel_pick<<<1, 1, 0, stream>>>(selst, b);
    }
    sel_gt<<<gdiv(n), 256, 0, stream>>>(scoreA, n, selst, perm);
    sel_eq<<<gdiv(n), 256, 0, stream>>>(scoreA, n, selst, perm, k);
    pool_x<<<gdiv((long long)k * 128), 256, 0, stream>>>(x, scoreA, perm, xp, xph, k);
    fill_u32<<<1, 128, 0, stream>>>(rmax, 0x007FFFFFu /* okey(-inf) */, 128);
    fill_f32<<<1, 128, 0, stream>>>(rsum, 0.f, 128);
    readout_acc<<<gdiv((long long)k * 128), 256, 0, stream>>>(xp, k, rmax, rsum);
    readout_add<<<1, 128, 0, stream>>>(rmax, rsum, (float)k, lbuf);
    if (mo) {
      fill_u32<<<gdiv(n), 256, 0, stream>>>((unsigned*)invA, 0xFFFFFFFFu, n);
      scatter_inv<<<gdiv(k), 256, 0, stream>>>(perm, k, invA);
      remap_edges<<<gdiv(E), 256, 0, stream>>>(ps, pd, pm, invA, ns, nd, mo, E);
    }
  };

  // ---- conv1: [x0[dst],x0[src],eattr] (44 -> pad 64) -> x1 [N,128], e1=eA
  fill_u32<<<gdiv((long long)N * 128), 256, 0, stream>>>(x1u, 0u, (long long)N * 128);
  edgeconv_wmma<false><<<512, 256, 128 * 64 * 2, stream>>>(
      x0h, 20, efh, 4, src, dst, nullptr, Wt1, 64, b1, eA, x1u, E);
  const float* x1 = (const float*)x1u;
  cvt_f32h<<<gdiv((long long)N * 128), 256, 0, stream>>>(x1, x1h, (long long)N * 128);

  // ---- conv2: [x1[dst],x1[src],e1] (384) -> x2, e2=eB
  fill_u32<<<gdiv((long long)N * 128), 256, 0, stream>>>(x2u, 0u, (long long)N * 128);
  edgeconv_wmma<true><<<512, 256, 128 * 384 * 2, stream>>>(
      x1h, 128, eA, 128, src, dst, nullptr, Wt2, 384, b2, eB, x2u, E);
  const float* x2 = (const float*)x2u;

  // ---- pool1 -> xp1 [k1,128] (+f16 mirror), edges (s2,d2,mask2), readout l1
  run_pool(x2, N, k1, src, dst, nullptr, gW2, ga2, gb2, xp1, xp1h, s2, d2, mask2);

  // ---- conv3 on pooled graph -> x3, e3=eA (reuse)
  fill_u32<<<gdiv((long long)k1 * 128), 256, 0, stream>>>(x3u, 0u, (long long)k1 * 128);
  edgeconv_wmma<true><<<512, 256, 128 * 384 * 2, stream>>>(
      xp1h, 128, eB, 128, s2, d2, mask2, Wt3, 384, b3, eA, x3u, E);
  const float* x3 = (const float*)x3u;

  // ---- pool2 -> xp2 [k2,128] (+f16 mirror), edges (s3,d3,mask3), readout l2
  run_pool(x3, k1, k2, s2, d2, mask2, gW3, ga3, gb3, xp2, xp2h, s3, d3, mask3);

  // ---- conv4 -> x4, e4=eB (reuse)
  fill_u32<<<gdiv((long long)k2 * 128), 256, 0, stream>>>(x4u, 0u, (long long)k2 * 128);
  edgeconv_wmma<true><<<512, 256, 128 * 384 * 2, stream>>>(
      xp2h, 128, eA, 128, s3, d3, mask3, Wt4, 384, b4, eB, x4u, E);
  const float* x4 = (const float*)x4u;

  // ---- pool3 -> xp3 [k3,128], readout l3 (no remap needed)
  run_pool(x4, k2, k3, s3, d3, mask3, gW4, ga4, gb4, xp3, xp3h, nullptr, nullptr, nullptr);

  // ---- head MLP: l -> [1,4]
  mlp_head<<<1, 256, 0, stream>>>(lbuf, lw1, lb1, lw2, lb2, lw3, lb3, (float*)d_out);
}
